// GNNModel_45122926411919
// MI455X (gfx1250) — matrix operand order repaired and checked by reading, not compiled, not run on hardware
//
#include <hip/hip_runtime.h>
#include <hip/hip_bf16.h>

// GNN (GraphNAS mixture) forward for MI455X (gfx1250, wave32, WMMA).
// Dense node GEMMs -> v_wmma_f32_16x16x32_f16 (f32 accum), packed-f16 B frags
// staged in LDS once per block. Edge aggregation -> single-pass float atomicAdd
// + order-encoded uint atomicMax.

typedef __attribute__((ext_vector_type(16))) _Float16 v16h;
typedef __attribute__((ext_vector_type(8)))  float    v8f;

#define H64 64

// ---------------------------------------------------------------------------
// Pack B[K,64] (f32, row-major) into WMMA B-fragment layout, f16:
//   fragment index = (kchunk*4 + coltile)*32 + lane ; 16 halves per fragment.
//   B-frag layout (16-bit B 32x16): elem j<8 -> K=8*(lane>=16)+j, j>=8 -> K=16+8*(lane>=16)+(j-8),
//   column N = lane&15 (+ 16*coltile).
// ---------------------------------------------------------------------------
__global__ void pack_B(const float* __restrict__ B, _Float16* __restrict__ Bp, int K)
{
    int i = blockIdx.x * blockDim.x + threadIdx.x;   // over K*64 halves
    if (i >= K * H64) return;
    int j    = i & 15;
    int lane = (i >> 4) & 31;
    int t    = (i >> 9) & 3;
    int c    = i >> 11;
    int k = c * 32 + ((lane < 16) ? 0 : 8) + ((j < 8) ? j : 8 + j);
    int n = t * 16 + (lane & 15);
    Bp[i] = (_Float16)B[k * H64 + n];
}

// ---------------------------------------------------------------------------
// WMMA GEMM: C[N,64] = concat_k(A0..A3)[N, NSEG*SEGK] * B + bias (+ReLU)
// One wave computes a 16x64 output tile; K loop fully unrolled; B in LDS.
// ---------------------------------------------------------------------------
template<int SEGK, int NSEG>
__global__ __launch_bounds__(256)
void gemm_wmma_f16(const float* __restrict__ A0, const float* __restrict__ A1,
                   const float* __restrict__ A2, const float* __restrict__ A3,
                   const _Float16* __restrict__ Bp, const float* __restrict__ bias,
                   float* __restrict__ C, int Nrows, int do_relu)
{
    constexpr int K   = SEGK * NSEG;
    constexpr int NCH = K / 32;

    // Stage the packed-B fragments in LDS once per block (<= 32KB).
    __shared__ __attribute__((aligned(32))) _Float16 Bs[K * H64];
    {
        const uint4* __restrict__ sv = (const uint4*)Bp;
        uint4* dv = (uint4*)Bs;
        constexpr int NV = K * H64 * 2 / 16;          // uint4 elements
        #pragma unroll
        for (int i = 0; i < (NV + 255) / 256; ++i) {
            int idx = i * 256 + threadIdx.x;
            if (NV % 256 == 0 || idx < NV) dv[idx] = sv[idx];
        }
    }
    __syncthreads();

    const int lane    = threadIdx.x & 31;
    const int wave    = threadIdx.x >> 5;
    const int rowTile = blockIdx.x * 8 + wave;
    const int nTiles  = (Nrows + 15) >> 4;
    if (rowTile >= nTiles) return;                    // after the barrier

    const float* segs[4] = {A0, A1, A2, A3};

    // A-fragment (16-bit A 16x32): lanes0-15 M=lane, elems0..7->K=0..7, 8..15->K=16..23
    //                              lanes16-31 M=lane-16, K offsets +8
    int arow = rowTile * 16 + (lane & 15);
    if (arow >= Nrows) arow = Nrows - 1;              // clamp (stores are guarded)
    const int aoff = (lane < 16) ? 0 : 8;
    const int bcol = lane & 15;

    v8f acc[4] = {};

    #pragma unroll
    for (int c = 0; c < NCH; ++c) {
        const float* ap = segs[(c * 32) / SEGK] + (size_t)arow * SEGK + (c * 32) % SEGK;
        const float4 x0 = *(const float4*)(ap + aoff);
        const float4 x1 = *(const float4*)(ap + aoff + 4);
        const float4 x2 = *(const float4*)(ap + 16 + aoff);
        const float4 x3 = *(const float4*)(ap + 16 + aoff + 4);
        v16h a;
        a[0]  = (_Float16)x0.x; a[1]  = (_Float16)x0.y; a[2]  = (_Float16)x0.z; a[3]  = (_Float16)x0.w;
        a[4]  = (_Float16)x1.x; a[5]  = (_Float16)x1.y; a[6]  = (_Float16)x1.z; a[7]  = (_Float16)x1.w;
        a[8]  = (_Float16)x2.x; a[9]  = (_Float16)x2.y; a[10] = (_Float16)x2.z; a[11] = (_Float16)x2.w;
        a[12] = (_Float16)x3.x; a[13] = (_Float16)x3.y; a[14] = (_Float16)x3.z; a[15] = (_Float16)x3.w;

        #pragma unroll
        for (int t = 0; t < 4; ++t) {
            const v16h b = *(const v16h*)(Bs + ((c * 4 + t) * 32 + lane) * 16);
            acc[t] = __builtin_amdgcn_wmma_f32_16x16x32_f16(
                false, a, false, b, (short)0, acc[t], false, false);
        }
    }

    // C/D layout: elem v -> M = v + (lane<16 ? 0 : 8), N = lane&15
    const int mbase = rowTile * 16 + ((lane < 16) ? 0 : 8);
    #pragma unroll
    for (int t = 0; t < 4; ++t) {
        const float bv = bias ? bias[t * 16 + bcol] : 0.0f;
        #pragma unroll
        for (int v = 0; v < 8; ++v) {
            const int row = mbase + v;
            if (row < Nrows) {
                float val = acc[t][v] + bv;
                if (do_relu) val = fmaxf(val, 0.0f);
                C[(size_t)row * H64 + t * 16 + bcol] = val;
            }
        }
    }
}

// ---------------------------------------------------------------------------
// Small elementwise / edge kernels
// ---------------------------------------------------------------------------
__global__ void fill_f32(float* p, float v, int n) {
    int i = blockIdx.x * blockDim.x + threadIdx.x;
    if (i < n) p[i] = v;
}
__global__ void fill_u32(unsigned* p, unsigned v, int n) {
    int i = blockIdx.x * blockDim.x + threadIdx.x;
    if (i < n) p[i] = v;
}
__global__ void count_edges(const int* __restrict__ tgt, float* __restrict__ cnt, int E) {
    int e = blockIdx.x * blockDim.x + threadIdx.x;
    if (e < E) atomicAdd(&cnt[tgt[e]], 1.0f);
}

__device__ __forceinline__ unsigned enc_f32(float f) {
    unsigned u = __float_as_uint(f);
    return (u & 0x80000000u) ? ~u : (u | 0x80000000u);   // monotonic order-preserving
}
__device__ __forceinline__ float dec_f32(unsigned k) {
    unsigned u = (k & 0x80000000u) ? (k & 0x7fffffffu) : ~k;
    return __uint_as_float(u);
}
#define ENC_NEG_INF 0x007FFFFFu   // enc(-inf)

// one wave32 per edge, 2 features per lane: float2 gather + atomic sum/max
__global__ __launch_bounds__(256)
void scatter_edges(const int* __restrict__ src, const int* __restrict__ tgt,
                   const float* __restrict__ xnl,
                   float* __restrict__ s, unsigned* __restrict__ mx, int E)
{
    int t = blockIdx.x * blockDim.x + threadIdx.x;
    int e = t >> 5;
    if (e >= E) return;
    int f  = (t & 31) * 2;
    int sv = src[e], tv = tgt[e];
    const float2 v = *(const float2*)(xnl + (size_t)sv * H64 + f);
    float*    sp = s  + (size_t)tv * H64 + f;
    unsigned* mp = mx + (size_t)tv * H64 + f;
    atomicAdd(sp,     v.x);
    atomicAdd(sp + 1, v.y);
    atomicMax(mp,     enc_f32(v.x));
    atomicMax(mp + 1, enc_f32(v.y));
}

// x_n = z0*sum + z1*mean + z2*max  (in place into s)
__global__ void agg_mix(float* __restrict__ s, const unsigned* __restrict__ mx,
                        const float* __restrict__ cnt, const float* __restrict__ Z, int n64)
{
    int i = blockIdx.x * blockDim.x + threadIdx.x;
    if (i >= n64) return;
    float z0 = Z[0], z1 = Z[1], z2 = Z[2];
    float sv = s[i];
    float c  = cnt[i >> 6];
    float mean = sv / fmaxf(c, 1.0f);
    float fmx  = (c > 0.0f) ? dec_f32(mx[i]) : 0.0f;
    s[i] = z0 * sv + z1 * mean + z2 * fmx;
}

// combine mixture + act mixture + LayerNorm; one wave32 per node (2 feats/lane)
__global__ __launch_bounds__(256)
void combine_act_ln(const float* __restrict__ xs, const float* __restrict__ xn,
                    const float* __restrict__ cc,
                    const float* __restrict__ Zc, const float* __restrict__ Za,
                    const float* __restrict__ pw,
                    const float* __restrict__ lng, const float* __restrict__ lnb,
                    float* __restrict__ out, int N)
{
    int lane = threadIdx.x & 31;
    int node = blockIdx.x * (blockDim.x >> 5) + (threadIdx.x >> 5);
    if (node >= N) return;
    float zc0 = Zc[0], zc1 = Zc[1], za0 = Za[0], za1 = Za[1], p = pw[0];
    size_t base = (size_t)node * H64;
    float h[2];
    #pragma unroll
    for (int j = 0; j < 2; ++j) {
        int f = lane + j * 32;
        float v = zc0 * (xs[base + f] + xn[base + f]) + zc1 * cc[base + f];
        h[j] = za0 * fmaxf(v, 0.0f) + za1 * ((v > 0.0f) ? v : p * v);
    }
    float sum = h[0] + h[1];
    #pragma unroll
    for (int o = 16; o > 0; o >>= 1) sum += __shfl_xor(sum, o, 32);
    float mu = sum * (1.0f / 64.0f);
    float d0 = h[0] - mu, d1 = h[1] - mu;
    float vs = d0 * d0 + d1 * d1;
    #pragma unroll
    for (int o = 16; o > 0; o >>= 1) vs += __shfl_xor(vs, o, 32);
    float inv = rsqrtf(vs * (1.0f / 64.0f) + 1e-5f);
    #pragma unroll
    for (int j = 0; j < 2; ++j) {
        int f = lane + j * 32;
        out[base + f] = (h[j] - mu) * inv * lng[f] + lnb[f];
    }
}

// x = z0*h + z1*(h+prev) + z2*lc_cat
__global__ void lc_mix(const float* __restrict__ h, const float* __restrict__ prev,
                       const float* __restrict__ lcc, const float* __restrict__ Z,
                       float* __restrict__ out, int n64)
{
    int i = blockIdx.x * blockDim.x + threadIdx.x;
    if (i >= n64) return;
    float z0 = Z[0], z1 = Z[1], z2 = Z[2];
    float hv = h[i];
    out[i] = z0 * hv + z1 * (hv + prev[i]) + z2 * lcc[i];
}

__global__ void max4(const float* __restrict__ a, const float* __restrict__ b,
                     const float* __restrict__ c, const float* __restrict__ d,
                     float* __restrict__ o, int n)
{
    int i = blockIdx.x * blockDim.x + threadIdx.x;
    if (i < n) o[i] = fmaxf(fmaxf(a[i], b[i]), fmaxf(c[i], d[i]));
}

__global__ void lagg_mix(const float* __restrict__ e3, const float* __restrict__ am,
                         const float* __restrict__ ac, const float* __restrict__ Z,
                         float* __restrict__ o, int n)
{
    int i = blockIdx.x * blockDim.x + threadIdx.x;
    if (i < n) o[i] = Z[0] * e3[i] + Z[1] * am[i] + Z[2] * ac[i];
}

// ---------------------------------------------------------------------------
// Orchestration
// ---------------------------------------------------------------------------
extern "C" void kernel_launch(void* const* d_in, const int* in_sizes, int n_in,
                              void* d_out, int out_size, void* d_ws, size_t ws_size,
                              hipStream_t stream)
{
    const float* x      = (const float*)d_in[0];
    const int*   ei     = (const int*)  d_in[1];
    const float* W_pre  = (const float*)d_in[2];
    const float* b_pre  = (const float*)d_in[3];
    const float* W_self = (const float*)d_in[4];
    const float* b_self = (const float*)d_in[5];
    const float* W_nbr  = (const float*)d_in[6];
    const float* b_nbr  = (const float*)d_in[7];
    const float* W_comb = (const float*)d_in[8];
    const float* b_comb = (const float*)d_in[9];
    const float* W_lc   = (const float*)d_in[10];
    const float* b_lc   = (const float*)d_in[11];
    const float* ln_g   = (const float*)d_in[12];
    const float* ln_b   = (const float*)d_in[13];
    const float* prelu  = (const float*)d_in[14];
    const float* W_lagg = (const float*)d_in[15];
    const float* b_lagg = (const float*)d_in[16];
    const float* W_ff1  = (const float*)d_in[17];
    const float* b_ff1  = (const float*)d_in[18];
    const float* W_ff2  = (const float*)d_in[19];
    const float* b_ff2  = (const float*)d_in[20];
    const float* Z_agg  = (const float*)d_in[21];
    const float* Z_comb = (const float*)d_in[22];
    const float* Z_act  = (const float*)d_in[23];
    const float* Z_lc   = (const float*)d_in[24];
    const float* Z_lagg = (const float*)d_in[25];

    const int N = in_sizes[0] / 128;   // 100000
    const int E = in_sizes[1] / 2;     // 1600000
    const int n64 = N * H64;
    const int L = 3;

    const int* src = ei;
    const int* tgt = ei + E;

    // workspace carve-up (floats; all offsets are 32B-aligned by construction)
    float* ws   = (float*)d_ws;
    const size_t nf = (size_t)n64;
    float* cnt  = ws;
    float* emb[4];
    emb[0] = cnt + N;
    emb[1] = emb[0] + nf;
    emb[2] = emb[1] + nf;
    emb[3] = emb[2] + nf;
    float* cur   = emb[3] + nf;
    float* xself = cur + nf;
    float* xnl   = xself + nf;
    float* sbuf  = xnl + nf;
    unsigned* mx = (unsigned*)(sbuf + nf);
    _Float16* Bp = (_Float16*)(mx + nf);          // 32KB packed-B staging

    const int TB = 256;
    const int gN64  = (n64 + TB - 1) / TB;
    const int gE    = (E + TB - 1) / TB;
    const int gEW   = (int)(((long long)E * 32 + TB - 1) / TB);
    const int gNode = (N + 7) / 8;                 // 8 nodes per 256-thread block
    const int gGemm = (((N + 15) / 16) + 7) / 8;   // 8 row-tiles per block

    #define PACK(Bsrc, K) pack_B<<<((K) * H64 + TB - 1) / TB, TB, 0, stream>>>((Bsrc), Bp, (K))

    // degree
    fill_f32<<<(N + TB - 1) / TB, TB, 0, stream>>>(cnt, 0.0f, N);
    count_edges<<<gE, TB, 0, stream>>>(tgt, cnt, E);

    // preprocess: emb0 = x @ W_pre + b_pre   (K=128)
    PACK(W_pre, 128);
    gemm_wmma_f16<128, 1><<<gGemm, TB, 0, stream>>>(x, nullptr, nullptr, nullptr,
                                                    Bp, b_pre, emb[0], N, 0);

    for (int i = 0; i < L; ++i) {
        const float* in = (i == 0) ? emb[0] : cur;
        // x_self / x_nl
        PACK(W_self + i * 4096, 64);
        gemm_wmma_f16<64, 1><<<gGemm, TB, 0, stream>>>(in, nullptr, nullptr, nullptr,
                                                       Bp, b_self + i * 64, xself, N, 0);
        PACK(W_nbr + i * 4096, 64);
        gemm_wmma_f16<64, 1><<<gGemm, TB, 0, stream>>>(in, nullptr, nullptr, nullptr,
                                                       Bp, b_nbr + i * 64, xnl, N, 0);
        // scatter sum + max
        fill_f32<<<gN64, TB, 0, stream>>>(sbuf, 0.0f, n64);
        fill_u32<<<gN64, TB, 0, stream>>>(mx, ENC_NEG_INF, n64);
        scatter_edges<<<gEW, TB, 0, stream>>>(src, tgt, xnl, sbuf, mx, E);
        agg_mix<<<gN64, TB, 0, stream>>>(sbuf, mx, cnt, Z_agg + i * 3, n64);
        // c_cat = [x_self | x_n] @ W_comb + b   (reuse xnl as output)
        PACK(W_comb + i * 8192, 128);
        gemm_wmma_f16<64, 2><<<gGemm, TB, 0, stream>>>(xself, sbuf, nullptr, nullptr,
                                                       Bp, b_comb + i * 64, xnl, N, 0);
        // combine + act + LN -> emb[i+1]
        combine_act_ln<<<gNode, TB, 0, stream>>>(xself, sbuf, xnl,
                                                 Z_comb + i * 2, Z_act + i * 2, prelu + i,
                                                 ln_g + i * 64, ln_b + i * 64, emb[i + 1], N);
        // lc_cat = [emb_i | emb_{i+1}] @ W_lc + b  (into xself)
        PACK(W_lc + i * 8192, 128);
        gemm_wmma_f16<64, 2><<<gGemm, TB, 0, stream>>>(emb[i], emb[i + 1], nullptr, nullptr,
                                                       Bp, b_lc + i * 64, xself, N, 0);
        lc_mix<<<gN64, TB, 0, stream>>>(emb[i + 1], emb[i], xself, Z_lc + i * 3, cur, n64);
    }

    // layer aggregation
    max4<<<gN64, TB, 0, stream>>>(emb[0], emb[1], emb[2], emb[3], sbuf, n64);
    PACK(W_lagg, 256);
    gemm_wmma_f16<64, 4><<<gGemm, TB, 0, stream>>>(emb[0], emb[1], emb[2], emb[3],
                                                   Bp, b_lagg, xnl, N, 0);
    lagg_mix<<<gN64, TB, 0, stream>>>(emb[3], sbuf, xnl, Z_lagg, xself, n64);

    // feed-forward head
    PACK(W_ff1, 64);
    gemm_wmma_f16<64, 1><<<gGemm, TB, 0, stream>>>(xself, nullptr, nullptr, nullptr,
                                                   Bp, b_ff1, sbuf, N, 1);
    PACK(W_ff2, 64);
    gemm_wmma_f16<64, 1><<<gGemm, TB, 0, stream>>>(sbuf, nullptr, nullptr, nullptr,
                                                   Bp, b_ff2, (float*)d_out, N, 0);

    #undef PACK
}
